// MultiHeadAttention_38843684225436
// MI455X (gfx1250) — matrix operand order repaired and checked by reading, not compiled, not run on hardware
//
#include <hip/hip_runtime.h>

// ---------------------------------------------------------------------------
// Problem constants (from reference): B=2, S=2048, D_MODEL=1024, H=16, DEPTH=64
// ---------------------------------------------------------------------------
#define BB      2
#define SS      2048
#define DM      1024
#define HH      16
#define DEPTH   64
#define ROWS    (BB * SS)          // 4096 flattened (b,s) rows
#define NEG_BIG (-1e9f)

typedef __attribute__((ext_vector_type(16))) __bf16 v16bf;
typedef __attribute__((ext_vector_type(8)))  float  v8f;

// ---------------------------------------------------------------------------
// Helpers
// ---------------------------------------------------------------------------
__device__ inline __bf16 f2bf(float f) {
  return (__bf16)f;   // native f32->bf16 convert on gfx1250
}

__device__ inline v8f wmma_bf16(v16bf a, v16bf b, v8f c) {
  // D = A(16x32 bf16) * B(32x16 bf16) + C(16x16 f32)
  return __builtin_amdgcn_wmma_f32_16x16x32_bf16(
      /*neg_a=*/false, a, /*neg_b=*/false, b,
      /*c_mod=*/(short)0, c, /*reuse_a=*/false, /*reuse_b=*/false);
}

// A-matrix 16x32 bf16 layout (ISA 7.12.2): lane m = lane&15, half hs = lane>>4.
// elements 0..7  -> K = kb + 8*hs + j        (VGPR0..3)
// elements 8..15 -> K = kb + 16 + 8*hs + j   (VGPR4..7)
__device__ inline v16bf load_a_bf(const __bf16* __restrict__ src, int ld,
                                  int rb, int kb, int lane) {
  int m = lane & 15, hs = lane >> 4;
  const __bf16* row = src + (size_t)(rb + m) * ld + kb;
  v16bf a;
#pragma unroll
  for (int j = 0; j < 8; ++j) a[j] = row[8 * hs + j];
#pragma unroll
  for (int j = 0; j < 8; ++j) a[8 + j] = row[16 + 8 * hs + j];
  return a;
}

// B = X^T where X is row-major [Nrows][K]: B(kk,n) = X[nb+n][kb+kk] -> contiguous.
__device__ inline v16bf load_b_transposed_bf(const __bf16* __restrict__ x, int ld,
                                             int kb, int nb, int lane) {
  int n = lane & 15, hs = lane >> 4;
  const __bf16* row = x + (size_t)(nb + n) * ld + kb;
  v16bf b;
#pragma unroll
  for (int j = 0; j < 16; ++j) b[j] = row[(j < 8) ? (8 * hs + j) : (16 + 8 * hs + (j - 8))];
  return b;
}

// Pre-packed B fragment: dst[((chunk*ntpm + ntile)*32 + lane)*16 + j]
// -> one contiguous 32-byte load per lane, fully coalesced across the wave.
__device__ inline v16bf load_b_frag(const __bf16* __restrict__ w, int ntpm,
                                    int chunk, int ntile, int lane) {
  const __bf16* p = w + ((size_t)(chunk * ntpm + ntile) * 32 + lane) * 16;
  return *(const v16bf*)p;
}

// ---------------------------------------------------------------------------
// 0) pre-pack a row-major f32 weight W[K][N] into WMMA B-fragment bf16 layout.
//    grid.y = matrix index (batched), 1 thread per destination element.
// ---------------------------------------------------------------------------
__global__ void prepack_b_kernel(const float* __restrict__ src,
                                 __bf16* __restrict__ dst,
                                 int K, int N, long srcStride, long dstStride) {
  int mat = blockIdx.y;
  int i = blockIdx.x * 256 + threadIdx.x;
  if (i >= K * N) return;
  const float* s = src + (size_t)mat * srcStride;
  __bf16* d = dst + (size_t)mat * dstStride;

  int j     = i & 15;
  int lane  = (i >> 4) & 31;
  int t     = i >> 9;
  int ntpm  = N >> 4;
  int ntile = t % ntpm;
  int chunk = t / ntpm;
  int hs = lane >> 4, n = lane & 15;
  int row = chunk * 32 + ((j < 8) ? (8 * hs + j) : (16 + 8 * hs + (j - 8)));
  int col = ntile * 16 + n;
  d[i] = f2bf(s[(size_t)row * N + col]);
}

// ---------------------------------------------------------------------------
// 1) projection chain: all 16 heads of q/k/v for a 16-row tile.
//    grid = (ROWS/16, 3), block = 32 (one wave).
//    qk: [t in {0,1}][head][row g][DEPTH] bf16 (row-major)
//    vt: [head][batch][DEPTH][S]           bf16 (transposed, for coalesced AV)
// ---------------------------------------------------------------------------
__global__ void proj_chain_kernel(const float* __restrict__ qin,
                                  const float* __restrict__ kin,
                                  const float* __restrict__ vin,
                                  const __bf16* __restrict__ wfirst_f, // frag [3]
                                  const __bf16* __restrict__ wrest_f,  // frag [3][15]
                                  __bf16* __restrict__ qk,
                                  __bf16* __restrict__ vt) {
  const int t    = blockIdx.y;
  const int rb   = blockIdx.x * 16;
  const int lane = threadIdx.x & 31;
  const int n    = lane & 15, hs = lane >> 4;
  const int m16  = lane & 15;

  const float*  src = (t == 0) ? qin : (t == 1) ? kin : vin;
  const __bf16* wf  = wfirst_f + (size_t)t * DM * DEPTH;
  const __bf16* wr  = wrest_f  + (size_t)t * (HH - 1) * DEPTH * DEPTH;

  const int b  = rb / SS;        // batch of this row tile
  const int sb = rb - b * SS;    // in-batch row base

  __shared__ float tile[16 * DEPTH];   // current head's 16x64 f32 result

  v8f acc[4];
#pragma unroll
  for (int nt = 0; nt < 4; ++nt) acc[nt] = (v8f)0.0f;

  // head 0: [16x1024] x [1024x64]
  for (int kc = 0; kc < DM; kc += 32) {
    // A chunk from f32 input
    const float* row = src + (size_t)(rb + m16) * DM + kc;
    v16bf a;
#pragma unroll
    for (int j = 0; j < 8; ++j) a[j] = f2bf(row[8 * hs + j]);
#pragma unroll
    for (int j = 0; j < 8; ++j) a[8 + j] = f2bf(row[16 + 8 * hs + j]);
#pragma unroll
    for (int nt = 0; nt < 4; ++nt)
      acc[nt] = wmma_bf16(a, load_b_frag(wf, 4, kc >> 5, nt, lane), acc[nt]);
  }

#pragma unroll 1
  for (int h = 0; h < HH; ++h) {
    // store head h result: LDS f32 (chain feed) + global bf16
#pragma unroll
    for (int nt = 0; nt < 4; ++nt)
#pragma unroll
      for (int r = 0; r < 8; ++r) {
        int m = r + 8 * hs;
        int col = nt * 16 + n;
        float v = acc[nt][r];
        tile[m * DEPTH + col] = v;
        if (t < 2) {
          __bf16* dsth = qk + ((size_t)(t * HH + h)) * ROWS * DEPTH;
          dsth[(size_t)(rb + m) * DEPTH + col] = f2bf(v);
        } else {
          __bf16* dsth = vt + ((size_t)(h * BB + b)) * DEPTH * SS;
          dsth[(size_t)col * SS + sb + m] = f2bf(v);   // transposed
        }
      }
    if (h == HH - 1) break;

    // next head: [16x64] x [64x64] from LDS tile (in-order DS per wave)
    const __bf16* wh = wr + (size_t)h * DEPTH * DEPTH;
#pragma unroll
    for (int nt = 0; nt < 4; ++nt) acc[nt] = (v8f)0.0f;
#pragma unroll
    for (int kc = 0; kc < DEPTH; kc += 32) {
      const float* trow = tile + m16 * DEPTH + kc;
      v16bf a;
#pragma unroll
      for (int j = 0; j < 8; ++j) a[j] = f2bf(trow[8 * hs + j]);
#pragma unroll
      for (int j = 0; j < 8; ++j) a[8 + j] = f2bf(trow[16 + 8 * hs + j]);
#pragma unroll
      for (int nt = 0; nt < 4; ++nt)
        acc[nt] = wmma_bf16(a, load_b_frag(wh, 4, kc >> 5, nt, lane), acc[nt]);
    }
  }
}

// ---------------------------------------------------------------------------
// 2) logits: Q Kᵀ * 1/sqrt(64) + causal mask, written straight into d_out.
//    One wave computes a 16x64 strip (4 key tiles) reusing each A chunk 4x.
//    grid = (S/64 key groups, B*H*(S/16) q tiles), block = 32.
//    attn index: (g*H + h)*S + k  with g = b*S + s  (== attns_concat layout)
// ---------------------------------------------------------------------------
__global__ void logits_kernel(const __bf16* __restrict__ qk,
                              float* __restrict__ attn) {
  const int ktg  = blockIdx.x;               // keys ktg*64 .. +63
  const int z    = blockIdx.y;
  const int qt   = z & (SS / 16 - 1);
  const int h    = (z >> 7) & (HH - 1);
  const int b    = z >> 11;
  const int lane = threadIdx.x & 31;
  const int n    = lane & 15, hs = lane >> 4;
  const int srow0 = qt * 16;

  if (ktg * 64 > srow0 + 15) {  // whole 16x64 strip causally masked
#pragma unroll
    for (int sub = 0; sub < 4; ++sub)
#pragma unroll
      for (int r = 0; r < 8; ++r) {
        int g = b * SS + srow0 + r + 8 * hs;
        attn[((size_t)g * HH + h) * SS + ktg * 64 + sub * 16 + n] = NEG_BIG;
      }
    return;
  }

  const __bf16* Q = qk + ((size_t)(0 * HH + h)) * ROWS * DEPTH + (size_t)b * SS * DEPTH;
  const __bf16* K = qk + ((size_t)(1 * HH + h)) * ROWS * DEPTH + (size_t)b * SS * DEPTH;

  v8f acc[4];
#pragma unroll
  for (int sub = 0; sub < 4; ++sub) acc[sub] = (v8f)0.0f;

#pragma unroll
  for (int kc = 0; kc < DEPTH; kc += 32) {
    v16bf a = load_a_bf(Q, DEPTH, srow0, kc, lane);
#pragma unroll
    for (int sub = 0; sub < 4; ++sub)
      acc[sub] = wmma_bf16(
          a, load_b_transposed_bf(K, DEPTH, kc, ktg * 64 + sub * 16, lane),
          acc[sub]);
  }

  const float scale = 0.125f;  // 1/sqrt(64)
#pragma unroll
  for (int sub = 0; sub < 4; ++sub)
#pragma unroll
    for (int r = 0; r < 8; ++r) {
      int m = r + 8 * hs;
      int srow = srow0 + m;
      int kcol = ktg * 64 + sub * 16 + n;
      float v = acc[sub][r] * scale + ((kcol > srow) ? NEG_BIG : 0.0f);
      int g = b * SS + srow;
      attn[((size_t)g * HH + h) * SS + kcol] = v;
    }
}

// ---------------------------------------------------------------------------
// 3) softmax in place over the last axis (k, length S). grid = B*S*H rows.
// ---------------------------------------------------------------------------
__global__ void softmax_kernel(float* __restrict__ attn) {
  const int rid = blockIdx.x;               // 0 .. B*S*H-1
  const int h   = rid & (HH - 1);
  const int g   = rid >> 4;                 // flattened b*S+s
  float* row = attn + ((size_t)g * HH + h) * SS;
  const int tid = threadIdx.x;              // 256 threads, 8 elems each

  __shared__ float red[256];

  float x[SS / 256];
  float mx = -3.4e38f;
#pragma unroll
  for (int i = 0; i < SS / 256; ++i) {
    x[i] = row[tid + 256 * i];
    mx = fmaxf(mx, x[i]);
  }
  red[tid] = mx;
  __syncthreads();
  for (int off = 128; off > 0; off >>= 1) {
    if (tid < off) red[tid] = fmaxf(red[tid], red[tid + off]);
    __syncthreads();
  }
  mx = red[0];
  __syncthreads();

  float sum = 0.0f;
#pragma unroll
  for (int i = 0; i < SS / 256; ++i) {
    x[i] = __expf(x[i] - mx);
    sum += x[i];
  }
  red[tid] = sum;
  __syncthreads();
  for (int off = 128; off > 0; off >>= 1) {
    if (tid < off) red[tid] += red[tid + off];
    __syncthreads();
  }
  float inv = 1.0f / red[0];

#pragma unroll
  for (int i = 0; i < SS / 256; ++i) row[tid + 256 * i] = x[i] * inv;
}

// ---------------------------------------------------------------------------
// 4) heads = attn @ V  -> heads_concat bf16 [ROWS][H*DEPTH]
//    V is stored transposed ([DEPTH][S]) so B loads are contiguous.
//    grid = (H, ROWS/16), block = 32.
// ---------------------------------------------------------------------------
__global__ void av_kernel(const float* __restrict__ attn,
                          const __bf16* __restrict__ vt,
                          __bf16* __restrict__ hc) {
  const int h    = blockIdx.x;
  const int qt   = blockIdx.y;
  const int lane = threadIdx.x & 31;
  const int n    = lane & 15, hs = lane >> 4;
  const int g0   = qt * 16;
  const int b    = g0 / SS;

  const __bf16* V = vt + ((size_t)(h * BB + b)) * DEPTH * SS;

  v8f acc[4];
#pragma unroll
  for (int nt = 0; nt < 4; ++nt) acc[nt] = (v8f)0.0f;

  const int m = lane & 15;
  const float* arow = attn + ((size_t)(g0 + m) * HH + h) * SS;

  for (int kc = 0; kc < SS; kc += 32) {
    if (kc + 64 < SS) __builtin_prefetch(arow + kc + 64, 0, 1);  // global_prefetch_b8
    v16bf a;
#pragma unroll
    for (int j = 0; j < 8; ++j) a[j] = f2bf(arow[kc + 8 * hs + j]);
#pragma unroll
    for (int j = 0; j < 8; ++j) a[8 + j] = f2bf(arow[kc + 16 + 8 * hs + j]);
#pragma unroll
    for (int nt = 0; nt < 4; ++nt)
      acc[nt] = wmma_bf16(a, load_b_transposed_bf(V, SS, kc, nt * 16, lane), acc[nt]);
  }

#pragma unroll
  for (int nt = 0; nt < 4; ++nt)
#pragma unroll
    for (int r = 0; r < 8; ++r) {
      int mm = r + 8 * hs;
      hc[(size_t)(g0 + mm) * (HH * DEPTH) + h * DEPTH + nt * 16 + n] = f2bf(acc[nt][r]);
    }
}

// ---------------------------------------------------------------------------
// 5) outputs = heads_concat @ w_h   [4096x1024] x [1024x1024] -> f32
//    One wave computes a 16x64 strip (4 N-tiles) reusing each A chunk 4x.
//    grid = (DM/64, ROWS/16), block = 32.
// ---------------------------------------------------------------------------
__global__ void final_gemm_kernel(const __bf16* __restrict__ hc,
                                  const __bf16* __restrict__ whf,
                                  float* __restrict__ out) {
  const int ng   = blockIdx.x;               // columns ng*64 .. +63
  const int qt   = blockIdx.y;
  const int lane = threadIdx.x & 31;
  const int n    = lane & 15, hs = lane >> 4;

  v8f acc[4];
#pragma unroll
  for (int j = 0; j < 4; ++j) acc[j] = (v8f)0.0f;

  for (int kc = 0; kc < DM; kc += 32) {
    v16bf a = load_a_bf(hc, DM, qt * 16, kc, lane);
    int chunk = kc >> 5;
#pragma unroll
    for (int j = 0; j < 4; ++j)
      acc[j] = wmma_bf16(a, load_b_frag(whf, DM / 16, chunk, ng * 4 + j, lane), acc[j]);
  }
#pragma unroll
  for (int j = 0; j < 4; ++j)
#pragma unroll
    for (int r = 0; r < 8; ++r) {
      int m = r + 8 * hs;
      out[(size_t)(qt * 16 + m) * DM + ng * 64 + j * 16 + n] = acc[j][r];
    }
}

// ---------------------------------------------------------------------------
// launch
// ---------------------------------------------------------------------------
extern "C" void kernel_launch(void* const* d_in, const int* in_sizes, int n_in,
                              void* d_out, int out_size, void* d_ws, size_t ws_size,
                              hipStream_t stream) {
  const float* q        = (const float*)d_in[0];
  const float* k        = (const float*)d_in[1];
  const float* v        = (const float*)d_in[2];
  // d_in[3] = mask (causal, recomputed from indices)
  const float* wq_first = (const float*)d_in[4];
  const float* wk_first = (const float*)d_in[5];
  const float* wv_first = (const float*)d_in[6];
  const float* wq_rest  = (const float*)d_in[7];
  const float* wk_rest  = (const float*)d_in[8];
  const float* wv_rest  = (const float*)d_in[9];
  const float* w_h      = (const float*)d_in[10];

  float* outputs = (float*)d_out;                      // [B,S,DM]
  float* attn    = outputs + (size_t)ROWS * DM;        // [B,S,H,S] == attns_concat

  // workspace carve-up (bf16 elements)
  __bf16* wfirst_f = (__bf16*)d_ws;                               // 3 * 1024*64
  __bf16* wrest_f  = wfirst_f + (size_t)3 * DM * DEPTH;           // 45 * 64*64
  __bf16* wh_f     = wrest_f + (size_t)3 * (HH - 1) * DEPTH * DEPTH;
  __bf16* qk       = wh_f + (size_t)DM * DM;                      // 2*16*4096*64
  __bf16* vt       = qk + (size_t)2 * HH * ROWS * DEPTH;          // 16*2*64*2048
  __bf16* hc       = vt + (size_t)HH * BB * DEPTH * SS;           // 4096*1024

  // 0) pre-pack all weights into WMMA B-fragment layout (bf16)
  {
    int nf = DM * DEPTH;      // 65536 per first-proj matrix
    prepack_b_kernel<<<dim3((nf + 255) / 256, 1), 256, 0, stream>>>(
        wq_first, wfirst_f + 0 * (size_t)nf, DM, DEPTH, 0, 0);
    prepack_b_kernel<<<dim3((nf + 255) / 256, 1), 256, 0, stream>>>(
        wk_first, wfirst_f + 1 * (size_t)nf, DM, DEPTH, 0, 0);
    prepack_b_kernel<<<dim3((nf + 255) / 256, 1), 256, 0, stream>>>(
        wv_first, wfirst_f + 2 * (size_t)nf, DM, DEPTH, 0, 0);

    int nr = DEPTH * DEPTH;   // 4096 per chain matrix, 15 per tensor
    prepack_b_kernel<<<dim3((nr + 255) / 256, HH - 1), 256, 0, stream>>>(
        wq_rest, wrest_f + 0 * (size_t)(HH - 1) * nr, DEPTH, DEPTH, nr, nr);
    prepack_b_kernel<<<dim3((nr + 255) / 256, HH - 1), 256, 0, stream>>>(
        wk_rest, wrest_f + 1 * (size_t)(HH - 1) * nr, DEPTH, DEPTH, nr, nr);
    prepack_b_kernel<<<dim3((nr + 255) / 256, HH - 1), 256, 0, stream>>>(
        wv_rest, wrest_f + 2 * (size_t)(HH - 1) * nr, DEPTH, DEPTH, nr, nr);

    int nh = DM * DM;         // 1M
    prepack_b_kernel<<<dim3((nh + 255) / 256, 1), 256, 0, stream>>>(
        w_h, wh_f, DM, DM, 0, 0);
  }

  // 1) head-chain projections (q,k for all heads; v stored transposed)
  proj_chain_kernel<<<dim3(ROWS / 16, 3), 32, 0, stream>>>(
      q, k, v, wfirst_f, wrest_f, qk, vt);

  // 2) masked scaled logits directly into d_out attn region
  logits_kernel<<<dim3(SS / 64, BB * HH * (SS / 16)), 32, 0, stream>>>(qk, attn);

  // 3) softmax in place
  softmax_kernel<<<BB * SS * HH, 256, 0, stream>>>(attn);

  // 4) attn @ V -> heads_concat (bf16)
  av_kernel<<<dim3(HH, ROWS / 16), 32, 0, stream>>>(attn, vt, hc);

  // 5) heads_concat @ w_h -> outputs
  final_gemm_kernel<<<dim3(DM / 64, ROWS / 16), 32, 0, stream>>>(hc, wh_f, outputs);
}
